// PointTransformerSeg_16750372454768
// MI455X (gfx1250) — matrix operand order repaired and checked by reading, not compiled, not run on hardware
//
#include <hip/hip_runtime.h>
#include <hip/hip_bf16.h>

// ---------------------------------------------------------------------------
// PointTransformerLayer forward for MI455X (gfx1250, wave32, WMMA).
//   N=65536 points, K=16 neighbors, C=64 channels, S=8 (C/S=8).
// Two kernels:
//   1) qkv_wmma : xq/xk/xv = x @ {Wq,Wk,Wv} + b   (WMMA f16->f32)
//   2) attn     : gather + pe + linear_w (two WMMA GEMMs) + softmax + reduce
// ---------------------------------------------------------------------------

#define NPTS 65536
#define KNN  16
#define CH   64
#define CS   8            // C / S

typedef __attribute__((ext_vector_type(16))) _Float16 v16h;
typedef __attribute__((ext_vector_type(8)))  float    v8f;
typedef __attribute__((ext_vector_type(2)))  _Float16 h2;

__constant__ constexpr float BNS = 0.99999500003749981f;  // 1/sqrt(1+1e-5)

// ===========================================================================
// Kernel 1: QKV projection.  One wave -> 16 rows x 64 cols of all 3 outputs.
// A fragment layout (16-bit A 16x32, ISA 7.12.2): lane m=lane&15 holds row M=m;
//   half h=lane>>4; VGPR v: K = (v<4 ? 2v : 16+2(v-4)) + 8h.
// B fragment (16-bit B 32x16): lane n=lane&15 holds col N=n;
//   VGPR v: K = (v<4 ? 2v : 16+2(v-4)) + 8h, same as A.
// D (f32 16x16): VGPR i holds M=i+8h, N=lane&15.
// ===========================================================================
__global__ __launch_bounds__(256) void qkv_wmma_kernel(
    const float* __restrict__ x,
    const float* __restrict__ Wq, const float* __restrict__ bq,
    const float* __restrict__ Wk, const float* __restrict__ bk,
    const float* __restrict__ Wv, const float* __restrict__ bv,
    float* __restrict__ oq, float* __restrict__ ok, float* __restrict__ ov)
{
  const int lane = threadIdx.x & 31;
  const int wave = threadIdx.x >> 5;
  const int r0   = (blockIdx.x * 8 + wave) * 16;     // 16-row tile per wave
  const int m    = lane & 15;
  const int h    = lane >> 4;

  // ---- A fragments: rows r0..r0+15 of x, K=0..63 (two 32-wide chunks) ----
  v16h a[2];
  const float* xrow = x + (size_t)(r0 + m) * CH;
  #pragma unroll
  for (int ch = 0; ch < 2; ++ch) {
    #pragma unroll
    for (int v = 0; v < 8; ++v) {
      const int kb = ch * 32 + (v < 4 ? 2 * v : 16 + 2 * (v - 4)) + 8 * h;
      a[ch][2 * v + 0] = (_Float16)xrow[kb + 0];
      a[ch][2 * v + 1] = (_Float16)xrow[kb + 1];
    }
  }

  const float* Ws[3]   = {Wq, Wk, Wv};
  const float* bs[3]   = {bq, bk, bv};
  float*       outs[3] = {oq, ok, ov};

  #pragma unroll
  for (int o = 0; o < 3; ++o) {
    const float* W = Ws[o];
    #pragma unroll
    for (int nt = 0; nt < 4; ++nt) {
      const int n = nt * 16 + m;                     // output column
      v8f acc = {};
      #pragma unroll
      for (int ch = 0; ch < 2; ++ch) {
        v16h b;
        #pragma unroll
        for (int v = 0; v < 8; ++v) {
          const int kb = ch * 32 + (v < 4 ? 2 * v : 16 + 2 * (v - 4)) + 8 * h;
          b[2 * v + 0] = (_Float16)W[(size_t)(kb + 0) * CH + n];
          b[2 * v + 1] = (_Float16)W[(size_t)(kb + 1) * CH + n];
        }
        acc = __builtin_amdgcn_wmma_f32_16x16x32_f16(
                  false, a[ch], false, b, (short)0, acc, false, false);
      }
      const float bias = bs[o][n];
      float* out = outs[o] + (size_t)r0 * CH + n;
      #pragma unroll
      for (int i = 0; i < 8; ++i)
        out[(size_t)(i + 8 * h) * CH] = acc[i] + bias;
    }
  }
}

// ===========================================================================
// Kernel 2: attention.  One wave per point; lane owns channels c0=lane,
// c1=lane+32.  Per-point 16x64 f16 "u" tile staged in LDS feeds two WMMA
// GEMMs (W1: K=64 in two chunks; W2: K padded).  Softmax over K in-lane.
// ===========================================================================
__global__ __launch_bounds__(256) void attn_kernel(
    const float* __restrict__ p,   const int* __restrict__ idx,
    const float* __restrict__ Wp1, const float* __restrict__ bp1,
    const float* __restrict__ gp,  const float* __restrict__ betap,
    const float* __restrict__ Wp2, const float* __restrict__ bp2,
    const float* __restrict__ g1,  const float* __restrict__ beta1,
    const float* __restrict__ W1,  const float* __restrict__ b1,
    const float* __restrict__ g2,  const float* __restrict__ beta2,
    const float* __restrict__ W2,  const float* __restrict__ b2,
    const float* __restrict__ xq,  const float* __restrict__ xk,
    const float* __restrict__ xv,  float* __restrict__ out)
{
  __shared__ _Float16 sW[8][KNN * CH];   // 8 waves * 2KB  : u tile (16x64 f16)
  __shared__ _Float16 sY[8][KNN * CS];   // 8 waves * 256B : bn2(relu) (16x8 f16)
  __shared__ float    sA[8][KNN * CS];   // 8 waves * 512B : softmax weights

  const int lane = threadIdx.x & 31;
  const int wave = threadIdx.x >> 5;
  const int i    = blockIdx.x * 8 + wave;            // point id
  const int c0   = lane, c1 = lane + 32;             // owned channels
  const int m    = lane & 15;                        // fragment row / col
  const int h    = lane >> 4;

  // ---- uniform small parameters (scalarized by compiler) ----
  float wp1[9], bp1r[3], gpr[3], bpr[3];
  #pragma unroll
  for (int t = 0; t < 9; ++t) wp1[t] = Wp1[t];
  #pragma unroll
  for (int t = 0; t < 3; ++t) {
    bp1r[t] = bp1[t]; gpr[t] = gp[t] * BNS; bpr[t] = betap[t];
  }

  // ---- per-lane channel parameters ----
  float wp2a[3], wp2b[3];
  #pragma unroll
  for (int t = 0; t < 3; ++t) { wp2a[t] = Wp2[t * CH + c0]; wp2b[t] = Wp2[t * CH + c1]; }
  const float bp2a = bp2[c0], bp2b = bp2[c1];
  const float g1a = g1[c0] * BNS, bt1a = beta1[c0];
  const float g1b = g1[c1] * BNS, bt1b = beta1[c1];

  // ---- B fragment for W1 (64x8, N padded to 16), two K chunks ----
  v16h bf1[2];
  #pragma unroll
  for (int ch = 0; ch < 2; ++ch) {
    #pragma unroll
    for (int v = 0; v < 8; ++v) {
      const int kb = ch * 32 + (v < 4 ? 2 * v : 16 + 2 * (v - 4)) + 8 * h;
      bf1[ch][2 * v + 0] = (m < CS) ? (_Float16)W1[(kb + 0) * CS + m] : (_Float16)0.f;
      bf1[ch][2 * v + 1] = (m < CS) ? (_Float16)W1[(kb + 1) * CS + m] : (_Float16)0.f;
    }
  }
  // ---- B fragment for W2 (8x8 -> K,N padded). K=0..7 -> lanes 0-15, v0..3 ----
  v16h bf2 = {};
  if (h == 0 && m < CS) {
    #pragma unroll
    for (int v = 0; v < 4; ++v) {
      bf2[2 * v + 0] = (_Float16)W2[(2 * v + 0) * CS + m];
      bf2[2 * v + 1] = (_Float16)W2[(2 * v + 1) * CS + m];
    }
  }
  // ---- per-N parameters for the middle of linear_w ----
  float b1n = 0.f, g2n = 0.f, be2n = 0.f, b2n = 0.f;
  if (m < CS) { b1n = b1[m]; g2n = g2[m] * BNS; be2n = beta2[m]; b2n = b2[m]; }

  // ---- per-point state ----
  const float xq0 = xq[(size_t)i * CH + c0];
  const float xq1 = xq[(size_t)i * CH + c1];
  const float pix = p[3 * i + 0], piy = p[3 * i + 1], piz = p[3 * i + 2];
  _Float16* myW = &sW[wave][0];
  _Float16* myY = &sY[wave][0];
  float*    myA = &sA[wave][0];

  // ======== pass 1: build u = relu(bn1(xk_g - xq + pe)) tile in LDS ========
  for (int k = 0; k < KNN; ++k) {
    const int j = idx[i * KNN + k];
    const float dx = p[3 * j + 0] - pix;
    const float dy = p[3 * j + 1] - piy;
    const float dz = p[3 * j + 2] - piz;
    float t0 = fmaf(dx, wp1[0], fmaf(dy, wp1[3], fmaf(dz, wp1[6], bp1r[0])));
    float t1 = fmaf(dx, wp1[1], fmaf(dy, wp1[4], fmaf(dz, wp1[7], bp1r[1])));
    float t2 = fmaf(dx, wp1[2], fmaf(dy, wp1[5], fmaf(dz, wp1[8], bp1r[2])));
    t0 = fmaxf(fmaf(t0, gpr[0], bpr[0]), 0.f);
    t1 = fmaxf(fmaf(t1, gpr[1], bpr[1]), 0.f);
    t2 = fmaxf(fmaf(t2, gpr[2], bpr[2]), 0.f);
    const float pe0 = fmaf(t0, wp2a[0], fmaf(t1, wp2a[1], fmaf(t2, wp2a[2], bp2a)));
    const float pe1 = fmaf(t0, wp2b[0], fmaf(t1, wp2b[1], fmaf(t2, wp2b[2], bp2b)));
    const float* xkj = xk + (size_t)j * CH;
    const float* xvj = xv + (size_t)j * CH;
    __builtin_prefetch(&xvj[c0], 0, 1);              // warm L2/L0 for pass 2
    const float w0 = xkj[c0] - xq0 + pe0;
    const float w1 = xkj[c1] - xq1 + pe1;
    myW[k * CH + c0] = (_Float16)fmaxf(fmaf(w0, g1a, bt1a), 0.f);
    myW[k * CH + c1] = (_Float16)fmaxf(fmaf(w1, g1b, bt1b), 0.f);
  }

  // ======== GEMM 1: y[16x8] = u[16x64] @ W1[64x8]  (two WMMA) ========
  v8f acc = {};
  #pragma unroll
  for (int ch = 0; ch < 2; ++ch) {
    v16h af;
    #pragma unroll
    for (int v = 0; v < 8; ++v) {
      const int kb = ch * 32 + (v < 4 ? 2 * v : 16 + 2 * (v - 4)) + 8 * h;
      const h2 hv = *(const h2*)&myW[m * CH + kb];
      af[2 * v + 0] = hv[0];
      af[2 * v + 1] = hv[1];
    }
    acc = __builtin_amdgcn_wmma_f32_16x16x32_f16(
              false, af, false, bf1[ch], (short)0, acc, false, false);
  }

  // ======== bn2 + relu, restage 16x8 tile for GEMM 2 ========
  if (m < CS) {
    #pragma unroll
    for (int ii = 0; ii < 8; ++ii) {
      const float y = acc[ii] + b1n;
      myY[(ii + 8 * h) * CS + m] = (_Float16)fmaxf(fmaf(y, g2n, be2n), 0.f);
    }
  }
  v16h a2 = {};
  if (h == 0) {
    #pragma unroll
    for (int v = 0; v < 4; ++v) {
      const h2 hv = *(const h2*)&myY[m * CS + 2 * v];
      a2[2 * v + 0] = hv[0];
      a2[2 * v + 1] = hv[1];
    }
  }
  v8f z = {};
  z = __builtin_amdgcn_wmma_f32_16x16x32_f16(
          false, a2, false, bf2, (short)0, z, false, false);

  // ======== softmax over K (M rows live in lane pair {n, n+16}) ========
  float zr[8];
  float mx = -1e30f;
  #pragma unroll
  for (int ii = 0; ii < 8; ++ii) { zr[ii] = z[ii] + b2n; mx = fmaxf(mx, zr[ii]); }
  mx = fmaxf(mx, __shfl_xor(mx, 16, 32));
  float s = 0.f;
  #pragma unroll
  for (int ii = 0; ii < 8; ++ii) { zr[ii] = __expf(zr[ii] - mx); s += zr[ii]; }
  s += __shfl_xor(s, 16, 32);
  const float inv = 1.0f / s;
  if (m < CS) {
    #pragma unroll
    for (int ii = 0; ii < 8; ++ii) myA[(ii + 8 * h) * CS + m] = zr[ii] * inv;
  }

  // ======== pass 2: out[i,c] = sum_k (xv_g + pe) * w[k, c%8] ========
  float o0 = 0.f, o1 = 0.f;
  const int rsel = lane & 7;
  for (int k = 0; k < KNN; ++k) {
    const int j = idx[i * KNN + k];
    const float dx = p[3 * j + 0] - pix;
    const float dy = p[3 * j + 1] - piy;
    const float dz = p[3 * j + 2] - piz;
    float t0 = fmaf(dx, wp1[0], fmaf(dy, wp1[3], fmaf(dz, wp1[6], bp1r[0])));
    float t1 = fmaf(dx, wp1[1], fmaf(dy, wp1[4], fmaf(dz, wp1[7], bp1r[1])));
    float t2 = fmaf(dx, wp1[2], fmaf(dy, wp1[5], fmaf(dz, wp1[8], bp1r[2])));
    t0 = fmaxf(fmaf(t0, gpr[0], bpr[0]), 0.f);
    t1 = fmaxf(fmaf(t1, gpr[1], bpr[1]), 0.f);
    t2 = fmaxf(fmaf(t2, gpr[2], bpr[2]), 0.f);
    const float pe0 = fmaf(t0, wp2a[0], fmaf(t1, wp2a[1], fmaf(t2, wp2a[2], bp2a)));
    const float pe1 = fmaf(t0, wp2b[0], fmaf(t1, wp2b[1], fmaf(t2, wp2b[2], bp2b)));
    const float* xvj = xv + (size_t)j * CH;
    const float wk = myA[k * CS + rsel];
    o0 = fmaf(xvj[c0] + pe0, wk, o0);
    o1 = fmaf(xvj[c1] + pe1, wk, o1);
  }
  out[(size_t)i * CH + c0] = o0;
  out[(size_t)i * CH + c1] = o1;
}

// ===========================================================================
// Launch: workspace holds xq | xk | xv (3 * 16 MB f32).
// ===========================================================================
extern "C" void kernel_launch(void* const* d_in, const int* in_sizes, int n_in,
                              void* d_out, int out_size, void* d_ws, size_t ws_size,
                              hipStream_t stream) {
  const float* p     = (const float*)d_in[0];
  const float* x     = (const float*)d_in[1];
  const int*   idx   = (const int*)  d_in[2];
  const float* Wq    = (const float*)d_in[3];
  const float* bq    = (const float*)d_in[4];
  const float* Wk    = (const float*)d_in[5];
  const float* bk    = (const float*)d_in[6];
  const float* Wv    = (const float*)d_in[7];
  const float* bv    = (const float*)d_in[8];
  const float* Wp1   = (const float*)d_in[9];
  const float* bp1   = (const float*)d_in[10];
  const float* gp    = (const float*)d_in[11];
  const float* betap = (const float*)d_in[12];
  const float* Wp2   = (const float*)d_in[13];
  const float* bp2   = (const float*)d_in[14];
  const float* g1    = (const float*)d_in[15];
  const float* beta1 = (const float*)d_in[16];
  const float* W1    = (const float*)d_in[17];
  const float* b1    = (const float*)d_in[18];
  const float* g2    = (const float*)d_in[19];
  const float* beta2 = (const float*)d_in[20];
  const float* W2    = (const float*)d_in[21];
  const float* b2    = (const float*)d_in[22];

  float* ws  = (float*)d_ws;
  float* xq  = ws;
  float* xkp = ws + (size_t)NPTS * CH;
  float* xvp = ws + 2 * (size_t)NPTS * CH;

  qkv_wmma_kernel<<<NPTS / 128, 256, 0, stream>>>(
      x, Wq, bq, Wk, bk, Wv, bv, xq, xkp, xvp);

  attn_kernel<<<NPTS / 8, 256, 0, stream>>>(
      p, idx, Wp1, bp1, gp, betap, Wp2, bp2,
      g1, beta1, W1, b1, g2, beta2, W2, b2,
      xq, xkp, xvp, (float*)d_out);
}
